// MultiHeadAttention_11836929868203
// MI455X (gfx1250) — matrix operand order repaired and checked by reading, not compiled, not run on hardware
//
#include <hip/hip_runtime.h>

typedef __bf16 bf16_t;
typedef __attribute__((ext_vector_type(16))) __bf16 v16bf;
typedef __attribute__((ext_vector_type(8)))  __bf16 v8bf;
typedef __attribute__((ext_vector_type(8)))  float  v8f;

constexpr int B_  = 2;
constexpr int S_  = 2048;
constexpr int D_  = 1024;
constexpr int H_  = 16;
constexpr int DK_ = 64;

// ---------- helpers ----------

__device__ inline bf16_t f2bf(float f) {
    union { float f; unsigned u; } x; x.f = f;
    unsigned r = x.u + 0x7FFFu + ((x.u >> 16) & 1u);   // round-to-nearest-even
    unsigned short h = (unsigned short)(r >> 16);
    return __builtin_bit_cast(bf16_t, h);
}

// 16-bit A-matrix 16x32 fragment (ISA 7.12.2):
// lanes 0-15 : row = lane,    K = 0..7  (v0-3) and 16..23 (v4-7)
// lanes16-31 : row = lane-16, K = 8..15 (v0-3) and 24..31 (v4-7)
__device__ inline v16bf load_a_frag(const bf16_t* base, int stride, int lane) {
    int r    = lane & 15;
    int koff = (lane >> 4) << 3;            // 0 or 8
    const bf16_t* p = base + (size_t)r * stride + koff;
    v8bf lo = *(const v8bf*)(p);            // K = koff .. koff+7
    v8bf hi = *(const v8bf*)(p + 16);       // K = koff+16 .. koff+23
    v16bf out;
#pragma unroll
    for (int i = 0; i < 8; ++i) { out[i] = lo[i]; out[i + 8] = hi[i]; }
    return out;
}

// 16-bit B-matrix 32x16 fragment:
// lanes 0-15 : N = lane,    K = 0..7  and 16..23
// lanes16-31 : N = lane-16, K = 8..15 and 24..31
// base points at element (K=0, N=0) of a [N][K]-major tile with row stride `stride`.
__device__ inline v16bf load_b_frag(const bf16_t* base, int stride, int lane) {
    int n    = lane & 15;
    int koff = (lane >> 4) << 3;            // 0 or 8
    const bf16_t* p = base + (size_t)n * stride + koff;
    v8bf lo = *(const v8bf*)(p);
    v8bf hi = *(const v8bf*)(p + 16);
    v16bf out;
#pragma unroll
    for (int i = 0; i < 8; ++i) { out[i] = lo[i]; out[i + 8] = hi[i]; }
    return out;
}

__device__ inline v8f wmma_bf16(v16bf a, v16bf b, v8f c) {
    return __builtin_amdgcn_wmma_f32_16x16x32_bf16(
        false, a, false, b, (short)0, c, false, false);
}

__device__ inline float halfmax(float v) {
#pragma unroll
    for (int s = 1; s < 16; s <<= 1) v = fmaxf(v, __shfl_xor(v, s, 32));
    return v;
}
__device__ inline float halfsum(float v) {
#pragma unroll
    for (int s = 1; s < 16; s <<= 1) v += __shfl_xor(v, s, 32);
    return v;
}

// ---------- kernels ----------

__global__ void k_cvt(const float* __restrict__ src, bf16_t* __restrict__ dst, int n) {
    int i = blockIdx.x * blockDim.x + threadIdx.x;
    if (i < n) dst[i] = f2bf(src[i]);
}

// C[M=4096, N=1024] = A[M,K=1024] * W[N,K]^T + bias, bf16 in / fp32 accum.
// Wave tile 32x64, K-loop software-pipelined 2 deep (unroll 64).
// mode 0: out bf16 [B,H,S,DK]   (Q, K projections)
// mode 1: out bf16 [B,H,DK,S]   (V projection, transposed for attention B-frags)
// mode 2: out fp32 row-major [M,N] (final projection -> d_out)
__global__ void __launch_bounds__(256)
k_gemm(const bf16_t* __restrict__ A, const bf16_t* __restrict__ W,
       const float* __restrict__ bias,
       bf16_t* __restrict__ outb, float* __restrict__ outf, int mode) {
    constexpr int K = D_;
    constexpr int N = D_;
    int lane = threadIdx.x & 31;
    int wave = threadIdx.x >> 5;
    int wm = blockIdx.y * 64  + (wave >> 2) * 32;   // 2 waves over M
    int wn = blockIdx.x * 256 + (wave & 3)  * 64;   // 4 waves over N

    const bf16_t* Arow0 = A + (size_t)wm * K;
    const bf16_t* Arow1 = A + (size_t)(wm + 16) * K;
    const bf16_t* Wrow[4];
#pragma unroll
    for (int i = 0; i < 4; ++i) Wrow[i] = W + (size_t)(wn + 16 * i) * K;

    const v8f vz = {};
    v8f acc[2][4];
#pragma unroll
    for (int r = 0; r < 2; ++r)
#pragma unroll
        for (int i = 0; i < 4; ++i) acc[r][i] = vz;

    // pipeline stage X: fragments for k
    v16bf xa0 = load_a_frag(Arow0, K, lane);
    v16bf xa1 = load_a_frag(Arow1, K, lane);
    v16bf xb0 = load_b_frag(Wrow[0], K, lane);
    v16bf xb1 = load_b_frag(Wrow[1], K, lane);
    v16bf xb2 = load_b_frag(Wrow[2], K, lane);
    v16bf xb3 = load_b_frag(Wrow[3], K, lane);

    for (int k = 0; k < K; k += 64) {
        // stage Y: fragments for k+32 (always valid: K % 64 == 0)
        v16bf ya0 = load_a_frag(Arow0 + k + 32, K, lane);
        v16bf ya1 = load_a_frag(Arow1 + k + 32, K, lane);
        v16bf yb0 = load_b_frag(Wrow[0] + k + 32, K, lane);
        v16bf yb1 = load_b_frag(Wrow[1] + k + 32, K, lane);
        v16bf yb2 = load_b_frag(Wrow[2] + k + 32, K, lane);
        v16bf yb3 = load_b_frag(Wrow[3] + k + 32, K, lane);
        __builtin_prefetch(Arow0 + k + 64, 0, 3);   // WGP-scope prefetch
        __builtin_prefetch(Wrow[0] + k + 64, 0, 3);

        acc[0][0] = wmma_bf16(xa0, xb0, acc[0][0]);
        acc[1][0] = wmma_bf16(xa1, xb0, acc[1][0]);
        acc[0][1] = wmma_bf16(xa0, xb1, acc[0][1]);
        acc[1][1] = wmma_bf16(xa1, xb1, acc[1][1]);
        acc[0][2] = wmma_bf16(xa0, xb2, acc[0][2]);
        acc[1][2] = wmma_bf16(xa1, xb2, acc[1][2]);
        acc[0][3] = wmma_bf16(xa0, xb3, acc[0][3]);
        acc[1][3] = wmma_bf16(xa1, xb3, acc[1][3]);

        if (k + 64 < K) {
            xa0 = load_a_frag(Arow0 + k + 64, K, lane);
            xa1 = load_a_frag(Arow1 + k + 64, K, lane);
            xb0 = load_b_frag(Wrow[0] + k + 64, K, lane);
            xb1 = load_b_frag(Wrow[1] + k + 64, K, lane);
            xb2 = load_b_frag(Wrow[2] + k + 64, K, lane);
            xb3 = load_b_frag(Wrow[3] + k + 64, K, lane);
            __builtin_prefetch(Arow1 + k + 96, 0, 3);
            __builtin_prefetch(Wrow[2] + k + 96, 0, 3);
        }

        acc[0][0] = wmma_bf16(ya0, yb0, acc[0][0]);
        acc[1][0] = wmma_bf16(ya1, yb0, acc[1][0]);
        acc[0][1] = wmma_bf16(ya0, yb1, acc[0][1]);
        acc[1][1] = wmma_bf16(ya1, yb1, acc[1][1]);
        acc[0][2] = wmma_bf16(ya0, yb2, acc[0][2]);
        acc[1][2] = wmma_bf16(ya1, yb2, acc[1][2]);
        acc[0][3] = wmma_bf16(ya0, yb3, acc[0][3]);
        acc[1][3] = wmma_bf16(ya1, yb3, acc[1][3]);
    }

    // C/D layout: lanes 0-15: vgpr j -> M=j; lanes 16-31: vgpr j -> M=j+8; N = lane&15
    int colL = lane & 15, rowoff = (lane >> 4) * 8;
#pragma unroll
    for (int r = 0; r < 2; ++r)
#pragma unroll
        for (int i = 0; i < 4; ++i) {
            int n = wn + 16 * i + colL;
            float bb = bias[n];
#pragma unroll
            for (int j = 0; j < 8; ++j) {
                int m = wm + 16 * r + j + rowoff;
                float v = acc[r][i][j] + bb;
                if (mode == 2) {
                    outf[(size_t)m * N + n] = v;
                } else {
                    int b = m >> 11, s = m & (S_ - 1);
                    int h = n >> 6,  dk = n & (DK_ - 1);
                    size_t idx = (mode == 0)
                        ? (((size_t)(b * H_ + h) * S_ + s) * DK_ + dk)
                        : (((size_t)(b * H_ + h) * DK_ + dk) * S_ + s);
                    outb[idx] = f2bf(v);
                }
            }
        }
}

// Flash-attention: one wave per 16-query tile of one (b,h); 32-key tiles,
// online softmax, P bounced through LDS to A-fragment layout, V pre-transposed.
__global__ void __launch_bounds__(128)
k_attn(const bf16_t* __restrict__ Q, const bf16_t* __restrict__ Km,
       const bf16_t* __restrict__ Vt, bf16_t* __restrict__ AO) {
    __shared__ bf16_t pbuf[4][16][32];
    int lane = threadIdx.x & 31;
    int wave = threadIdx.x >> 5;
    int tile = blockIdx.x * 4 + wave;       // 0 .. B*H*(S/16)-1 = 4095
    int b  = tile >> 11;
    int h  = (tile >> 7) & 15;
    int qt = tile & 127;
    int qrow0 = qt * 16;
    int bh = b * H_ + h;
    int colL = lane & 15, rowoff = (lane >> 4) * 8;

    const bf16_t* qbase = Q + ((size_t)bh * S_ + qrow0) * DK_;
    v16bf aq0 = load_a_frag(qbase,      DK_, lane);   // features 0..31
    v16bf aq1 = load_a_frag(qbase + 32, DK_, lane);   // features 32..63

    const v8f vz = {};
    float m[8], l[8];
    v8f o[4];
#pragma unroll
    for (int j = 0; j < 8; ++j) { m[j] = -1e30f; l[j] = 0.f; }
#pragma unroll
    for (int c = 0; c < 4; ++c) o[c] = vz;

    int qlast = qrow0 + 15;
    for (int kb = 0; kb <= qlast; kb += 32) {
        const bf16_t* kb0 = Km + ((size_t)bh * S_ + kb) * DK_;
        v16bf b00 = load_b_frag(kb0,                 DK_, lane); // keys kb..+15,  f 0..31
        v16bf b01 = load_b_frag(kb0 + 32,            DK_, lane); // keys kb..+15,  f 32..63
        v16bf b10 = load_b_frag(kb0 + 16 * DK_,      DK_, lane); // keys kb+16..,  f 0..31
        v16bf b11 = load_b_frag(kb0 + 16 * DK_ + 32, DK_, lane); // keys kb+16..,  f 32..63
        v8f s0 = vz, s1 = vz;
        s0 = wmma_bf16(aq0, b00, s0);
        s0 = wmma_bf16(aq1, b01, s0);
        s1 = wmma_bf16(aq0, b10, s1);
        s1 = wmma_bf16(aq1, b11, s1);

        float scl[8];
#pragma unroll
        for (int j = 0; j < 8; ++j) {
            int row = qrow0 + j + rowoff;
            float a0 = s0[j] * 0.125f;            // 1/sqrt(64)
            float a1 = s1[j] * 0.125f;
            if (kb + colL > row)      a0 = -1e30f;    // causal mask
            if (kb + 16 + colL > row) a1 = -1e30f;
            float rm = halfmax(fmaxf(a0, a1));
            float mn = fmaxf(m[j], rm);
            float sc = __expf(m[j] - mn);
            float p0 = __expf(a0 - mn);
            float p1 = __expf(a1 - mn);
            l[j] = l[j] * sc + halfsum(p0 + p1);
            m[j] = mn; scl[j] = sc;
            pbuf[wave][j + rowoff][colL]      = f2bf(p0);
            pbuf[wave][j + rowoff][16 + colL] = f2bf(p1);
        }
#pragma unroll
        for (int c = 0; c < 4; ++c)
#pragma unroll
            for (int j = 0; j < 8; ++j) o[c][j] *= scl[j];

        // P (16x32) back out of LDS in A-fragment layout; in-order DS within wave.
        v16bf pf = load_a_frag(&pbuf[wave][0][0], 32, lane);
        const bf16_t* vb = Vt + ((size_t)bh * DK_) * S_ + kb;
#pragma unroll
        for (int c = 0; c < 4; ++c) {
            v16bf bv = load_b_frag(vb + (size_t)(c * 16) * S_, S_, lane);
            o[c] = wmma_bf16(pf, bv, o[c]);
        }
    }

    // normalize and store [B, S, H, DK]
#pragma unroll
    for (int j = 0; j < 8; ++j) {
        float inv = 1.f / l[j];
        int row = qrow0 + j + rowoff;
        size_t base = ((size_t)(b * S_ + row)) * D_ + h * DK_;
#pragma unroll
        for (int c = 0; c < 4; ++c)
            AO[base + c * 16 + colL] = f2bf(o[c][j] * inv);
    }
}

// ---------- launch ----------

extern "C" void kernel_launch(void* const* d_in, const int* in_sizes, int n_in,
                              void* d_out, int out_size, void* d_ws, size_t ws_size,
                              hipStream_t stream) {
    (void)in_sizes; (void)n_in; (void)out_size; (void)ws_size;
    const float* x  = (const float*)d_in[0];
    // d_in[1] = causal mask: computed analytically, unused
    const float* wq = (const float*)d_in[2];
    const float* bq = (const float*)d_in[3];
    const float* wk = (const float*)d_in[4];
    const float* bk = (const float*)d_in[5];
    const float* wv = (const float*)d_in[6];
    const float* bv = (const float*)d_in[7];
    const float* wo = (const float*)d_in[8];
    const float* bo = (const float*)d_in[9];
    float* out = (float*)d_out;

    const size_t nX = (size_t)B_ * S_ * D_;   // 4,194,304
    const size_t nW = (size_t)D_ * D_;        // 1,048,576
    bf16_t* ws  = (bf16_t*)d_ws;
    bf16_t* xb  = ws;
    bf16_t* wqb = xb  + nX;
    bf16_t* wkb = wqb + nW;
    bf16_t* wvb = wkb + nW;
    bf16_t* wob = wvb + nW;
    bf16_t* Qb  = wob + nW;
    bf16_t* Kb  = Qb  + nX;
    bf16_t* Vt  = Kb  + nX;
    bf16_t* AO  = Vt  + nX;   // end: ~48 MB of d_ws

    k_cvt<<<(int)((nX + 255) / 256), 256, 0, stream>>>(x,  xb,  (int)nX);
    k_cvt<<<(int)((nW + 255) / 256), 256, 0, stream>>>(wq, wqb, (int)nW);
    k_cvt<<<(int)((nW + 255) / 256), 256, 0, stream>>>(wk, wkb, (int)nW);
    k_cvt<<<(int)((nW + 255) / 256), 256, 0, stream>>>(wv, wvb, (int)nW);
    k_cvt<<<(int)((nW + 255) / 256), 256, 0, stream>>>(wo, wob, (int)nW);

    dim3 gg(D_ / 256, (B_ * S_) / 64);        // (4, 64)
    k_gemm<<<gg, 256, 0, stream>>>(xb, wqb, bq, Qb, nullptr, 0);
    k_gemm<<<gg, 256, 0, stream>>>(xb, wkb, bk, Kb, nullptr, 0);
    k_gemm<<<gg, 256, 0, stream>>>(xb, wvb, bv, Vt, nullptr, 1);

    k_attn<<<(B_ * H_ * (S_ / 16)) / 4, 128, 0, stream>>>(Qb, Kb, Vt, AO);

    k_gemm<<<gg, 256, 0, stream>>>(AO, wob, bo, nullptr, out, 2);
}
